// MANN_35527969472749
// MI455X (gfx1250) — matrix operand order repaired and checked by reading, not compiled, not run on hardware
//
#include <hip/hip_runtime.h>
#include <hip/hip_bf16.h>

typedef unsigned short ushort_t;
typedef __attribute__((ext_vector_type(16))) __bf16        v16bf;
typedef __attribute__((ext_vector_type(16))) unsigned short v16us;
typedef __attribute__((ext_vector_type(8)))  unsigned short us8;
typedef __attribute__((ext_vector_type(8)))  float          v8f;

// ---------------- constants ----------------
#define BATCH   512
#define T_SEQ   55
#define M1      (BATCH * T_SEQ)   // 28160
#define KP1     800               // 789 padded to 32-multiple
#define N1      2048              // 4*H
#define HID     512
#define N2P     64                // 20 padded

// ---------------- helpers ----------------
__device__ __forceinline__ unsigned short f2bf(float f) {
    unsigned int u = __float_as_uint(f);
    unsigned int r = (u + 0x7FFFu + ((u >> 16) & 1u)) >> 16;
    return (unsigned short)r;
}
__device__ __forceinline__ v16bf pk16(us8 lo, us8 hi) {
    v16us u = __builtin_shufflevector(lo, hi, 0,1,2,3,4,5,6,7,8,9,10,11,12,13,14,15);
    return __builtin_bit_cast(v16bf, u);
}
__device__ __forceinline__ v8f v8f_zero() {
    v8f v;
#pragma unroll
    for (int i = 0; i < 8; ++i) v[i] = 0.f;
    return v;
}
__device__ __forceinline__ float sigf(float x) { return 1.0f / (1.0f + expf(-x)); }

// CDNA5 async memory->LDS copy (ASYNCcnt-tracked, no VGPR round-trip).
// VDST = per-lane LDS byte address, VADDR = per-lane 64-bit global address.
__device__ __forceinline__ unsigned lds_off_u32(const void* p) {
    return (unsigned)(unsigned long long)(uintptr_t)p;   // low 32 bits = LDS address
}
__device__ __forceinline__ void async_g2l_b128(unsigned lds_off, const void* g) {
    asm volatile("global_load_async_to_lds_b128 %0, %1, off"
                 :: "v"(lds_off), "v"(g) : "memory");
}
__device__ __forceinline__ void wait_async0() {
    asm volatile("s_wait_asynccnt 0x0" ::: "memory");
}

// ---------------- pack kernels ----------------
// X bf16 [M1][KP1]: concat(images, labels(or 0 for query)), zero-padded K
__global__ void pack_x_kernel(const float* __restrict__ imgs,
                              const float* __restrict__ lbls,
                              ushort_t* __restrict__ X) {
    long long idx = (long long)blockIdx.x * blockDim.x + threadIdx.x;
    if (idx >= (long long)M1 * KP1) return;
    int m = (int)(idx / KP1), k = (int)(idx % KP1);
    int b = m / T_SEQ, t = m % T_SEQ;
    int k1 = t / 5, n = t % 5;
    float v = 0.f;
    if (k < 784)
        v = imgs[(((size_t)b * 11 + k1) * 5 + n) * 784 + k];
    else if (k < 789 && k1 < 10)
        v = lbls[(((size_t)b * 11 + k1) * 5 + n) * 5 + (k - 784)];
    X[idx] = f2bf(v);
}

// W1 f32[789][2048] -> bf16[800][2048] zero-padded rows
__global__ void pack_w1_kernel(const float* __restrict__ W, ushort_t* __restrict__ Wp) {
    long long idx = (long long)blockIdx.x * blockDim.x + threadIdx.x;
    if (idx >= (long long)KP1 * N1) return;
    int r = (int)(idx / N1), c = (int)(idx % N1);
    Wp[idx] = (r < 789) ? f2bf(W[(size_t)r * N1 + c]) : (unsigned short)0;
}

// U1 f32[512][2048] -> bf16 packed in WMMA B-fragment order:
// offset = ((nt*16 + kt)*32 + lane)*16 + j ; K = kt*32 + 16*(lane>>4) + j ; N = nt*16 + (lane&15)
__global__ void pack_u1_kernel(const float* __restrict__ U, ushort_t* __restrict__ Up) {
    int idx = blockIdx.x * blockDim.x + threadIdx.x;
    if (idx >= 128 * 16 * 32 * 16) return;
    int j    =  idx        & 15;
    int lane = (idx >> 4)  & 31;
    int kt   = (idx >> 9)  & 15;
    int nt   =  idx >> 13;
    int K = kt * 32 + 16 * (lane >> 4) + j;
    int N = nt * 16 + (lane & 15);
    Up[idx] = f2bf(U[(size_t)K * N1 + N]);
}

// W2 f32[512][20] -> bf16[512][64] zero-padded cols
__global__ void pack_w2_kernel(const float* __restrict__ W, ushort_t* __restrict__ Wp) {
    int idx = blockIdx.x * blockDim.x + threadIdx.x;
    if (idx >= HID * N2P) return;
    int r = idx / N2P, c = idx % N2P;
    Wp[idx] = (c < 20) ? f2bf(W[r * 20 + c]) : (unsigned short)0;
}
__global__ void pack_b2_kernel(const float* __restrict__ b, float* __restrict__ bp) {
    int c = threadIdx.x;
    if (c < N2P) bp[c] = (c < 20) ? b[c] : 0.f;
}

// ---------------- generic bf16 WMMA GEMM:  C = A*B + bias ----------------
// tile 64x64, K-step 32, 128 threads (4 waves); wave w -> rows w*16..+16, all 64 cols
// A tile staged memory->LDS with GLOBAL_LOAD_ASYNC_TO_LDS_B128 (ASYNCcnt),
// B tile staged manually because it is transposed into LDS.
#define GSTR 40   // LDS row stride (ushorts), multiple of 8
__global__ __launch_bounds__(128) void gemm_bf16_kernel(
    const ushort_t* __restrict__ A, const ushort_t* __restrict__ B,
    const float* __restrict__ bias, float* __restrict__ C,
    int K, int lda, int ldb, int ldc, int remap) {
    __shared__ ushort_t As[64 * GSTR];
    __shared__ ushort_t Bt[64 * GSTR];
    const int tid  = threadIdx.x;
    const int w    = tid >> 5, lane = tid & 31;
    const int g    = lane >> 4, lm = lane & 15;
    const int m0   = blockIdx.x * 64;
    const int n0   = blockIdx.y * 64;

    v8f acc[4];
#pragma unroll
    for (int s = 0; s < 4; ++s) acc[s] = v8f_zero();

    const int ar  = tid >> 1;          // 0..63
    const int ac  = (tid & 1) * 16;    // 0,16
    const int bkr = tid >> 2;          // 0..31
    const int bc  = (tid & 3) * 16;    // 0..48

    const unsigned as_lds0 = lds_off_u32(&As[ar * GSTR + ac]);
    const unsigned as_lds1 = lds_off_u32(&As[ar * GSTR + ac + 8]);

    for (int k0 = 0; k0 < K; k0 += 32) {
        __syncthreads();
        // --- A tile: async global->LDS (no VGPR round-trip) ---
        const ushort_t* ap = A + (size_t)(m0 + ar) * lda + k0 + ac;
        async_g2l_b128(as_lds0, ap);
        async_g2l_b128(as_lds1, ap + 8);

        // --- B tile: load + transpose into LDS ---
        const ushort_t* bp = B + (size_t)(k0 + bkr) * ldb + n0 + bc;
        us8 b0 = *(const us8*)bp;
        us8 b1 = *(const us8*)(bp + 8);
        if (k0 + 32 < K) {
            __builtin_prefetch((const void*)(ap + 32), 0, 1);
            __builtin_prefetch((const void*)(bp + (size_t)32 * ldb), 0, 1);
        }
#pragma unroll
        for (int i = 0; i < 8; ++i) {
            Bt[(bc + i) * GSTR + bkr]     = b0[i];
            Bt[(bc + 8 + i) * GSTR + bkr] = b1[i];
        }
        wait_async0();       // async LDS writes visible before the barrier
        __syncthreads();

        const int abase = (w * 16 + lm) * GSTR + 8 * g;
        us8 alo = *(const us8*)&As[abase];
        us8 ahi = *(const us8*)&As[abase + 16];
        v16bf af = pk16(alo, ahi);
#pragma unroll
        for (int s = 0; s < 4; ++s) {
            const int bbase = (s * 16 + lm) * GSTR + 8 * g;
            us8 blo = *(const us8*)&Bt[bbase];
            us8 bhi = *(const us8*)&Bt[bbase + 16];
            v16bf bf = pk16(blo, bhi);
            acc[s] = __builtin_amdgcn_wmma_f32_16x16x32_bf16(
                false, af, false, bf, (short)0, acc[s], false, false);
        }
    }
#pragma unroll
    for (int s = 0; s < 4; ++s) {
        const int col = n0 + s * 16 + lm;
        const float bv = bias[col];
#pragma unroll
        for (int r = 0; r < 8; ++r) {
            int m  = m0 + w * 16 + r + 8 * g;
            int mo = remap ? ((m % T_SEQ) * BATCH + m / T_SEQ) : m;
            C[(size_t)mo * ldc + col] = acc[s][r] + bv;
        }
    }
}

// ---------------- LSTM layer 1: persistent batch-parallel recurrence ----------------
// 32 blocks x 256 threads; block owns 16 batch rows; wave w owns gate columns [w*64, w*64+64)
// and computes z_{i,f,g,o} strips itself -> gate math stays in registers.
#define HSTR 528  // h LDS stride (ushorts), 16B-aligned multiple
__global__ __launch_bounds__(256) void lstm1_kernel(
    const ushort_t* __restrict__ Upack,   // prepacked U1 bf16 fragments
    const float*    __restrict__ xw1t,    // [55][512][2048] f32
    ushort_t*       __restrict__ x1bf) {  // [28160][512] bf16
    __shared__ ushort_t hbuf[16 * HSTR];
    const int tid  = threadIdx.x;
    const int w    = tid >> 5, lane = tid & 31;
    const int g    = lane >> 4, lm = lane & 15;
    const int r0   = blockIdx.x * 16;

    for (int i = tid; i < 16 * HSTR; i += 256) hbuf[i] = 0;
    __syncthreads();

    v8f c_reg[4];
#pragma unroll
    for (int s = 0; s < 4; ++s) c_reg[s] = v8f_zero();

    for (int t = 0; t < T_SEQ; ++t) {
        v8f acc[4][4];
#pragma unroll
        for (int gi = 0; gi < 4; ++gi)
#pragma unroll
            for (int s = 0; s < 4; ++s) acc[gi][s] = v8f_zero();

#pragma unroll 1
        for (int kt = 0; kt < 16; ++kt) {
            const int abase = lm * HSTR + kt * 32 + 16 * g;
            us8 alo = *(const us8*)&hbuf[abase];
            us8 ahi = *(const us8*)&hbuf[abase + 8];
            v16bf af = pk16(alo, ahi);
#pragma unroll
            for (int gi = 0; gi < 4; ++gi) {
#pragma unroll
                for (int s = 0; s < 4; ++s) {
                    const int nt = gi * 32 + w * 4 + s;
                    const ushort_t* bp = Upack + (((size_t)nt * 16 + kt) * 32 + lane) * 16;
                    us8 blo = *(const us8*)bp;
                    us8 bhi = *(const us8*)(bp + 8);
                    v16bf bf = pk16(blo, bhi);
                    acc[gi][s] = __builtin_amdgcn_wmma_f32_16x16x32_bf16(
                        false, af, false, bf, (short)0, acc[gi][s], false, false);
                }
            }
        }
        __syncthreads();  // all waves done reading h before we overwrite it

        const float* zrow = xw1t + ((size_t)t * BATCH + r0) * N1;
#pragma unroll
        for (int s = 0; s < 4; ++s) {
            const int j = w * 64 + s * 16 + lm;  // hidden column
#pragma unroll
            for (int r = 0; r < 8; ++r) {
                const int bl = r + 8 * g;        // batch row within block
                const float* zp = zrow + (size_t)bl * N1 + j;
                float zi = acc[0][s][r] + zp[0];
                float zf = acc[1][s][r] + zp[512];
                float zg = acc[2][s][r] + zp[1024];
                float zo = acc[3][s][r] + zp[1536];
                float ig = sigf(zi), fg = sigf(zf);
                float gg = tanhf(zg), og = sigf(zo);
                float cc = fg * c_reg[s][r] + ig * gg;
                c_reg[s][r] = cc;
                float hh = og * tanhf(cc);
                unsigned short hb = f2bf(hh);
                hbuf[bl * HSTR + j] = hb;
                x1bf[((size_t)(r0 + bl) * T_SEQ + t) * HID + j] = hb;
            }
        }
        __syncthreads();  // h fully updated before next step reads
    }
}

// ---------------- LSTM layer 2: one thread per batch row ----------------
__global__ __launch_bounds__(128) void lstm2_kernel(
    const float* __restrict__ xw2,   // [28160][64] f32 (cols 0..19 valid)
    const float* __restrict__ U2,    // [5][20] f32
    float* __restrict__ out) {       // [B,55,5] == [B,11,5,5] flat
    __shared__ float U2s[100];
    const int tid = threadIdx.x;
    if (tid < 100) U2s[tid] = U2[tid];
    __syncthreads();
    const int b = blockIdx.x * 128 + tid;
    float h[5], c[5];
#pragma unroll
    for (int j = 0; j < 5; ++j) { h[j] = 0.f; c[j] = 0.f; }
    for (int t = 0; t < T_SEQ; ++t) {
        const float* xw = xw2 + ((size_t)b * T_SEQ + t) * N2P;
        float z[20];
#pragma unroll
        for (int j = 0; j < 20; ++j) z[j] = xw[j];
#pragma unroll
        for (int k = 0; k < 5; ++k) {
            float hk = h[k];
#pragma unroll
            for (int j = 0; j < 20; ++j) z[j] += hk * U2s[k * 20 + j];
        }
        float* orow = out + ((size_t)b * T_SEQ + t) * 5;
#pragma unroll
        for (int j = 0; j < 5; ++j) {
            float ig = sigf(z[j]);
            float fg = sigf(z[5 + j]);
            float gg = tanhf(z[10 + j]);
            float og = sigf(z[15 + j]);
            c[j] = fg * c[j] + ig * gg;
            h[j] = og * tanhf(c[j]);
            orow[j] = h[j];
        }
    }
}

// ---------------- launch ----------------
extern "C" void kernel_launch(void* const* d_in, const int* in_sizes, int n_in,
                              void* d_out, int out_size, void* d_ws, size_t ws_size,
                              hipStream_t stream) {
    const float* imgs = (const float*)d_in[0];
    const float* lbls = (const float*)d_in[1];
    const float* W1   = (const float*)d_in[2];
    const float* U1   = (const float*)d_in[3];
    const float* b1   = (const float*)d_in[4];
    const float* W2   = (const float*)d_in[5];
    const float* U2   = (const float*)d_in[6];
    const float* b2   = (const float*)d_in[7];
    float* out = (float*)d_out;

    char* ws = (char*)d_ws;
    size_t off = 0;
    auto carve = [&](size_t bytes) -> char* {
        char* p = ws + off;
        off = (off + bytes + 255) & ~(size_t)255;
        return p;
    };
    ushort_t* X    = (ushort_t*)carve((size_t)M1 * KP1 * 2);
    ushort_t* Wp1  = (ushort_t*)carve((size_t)KP1 * N1 * 2);
    ushort_t* Up1  = (ushort_t*)carve((size_t)128 * 16 * 32 * 16 * 2);
    ushort_t* Wp2  = (ushort_t*)carve((size_t)HID * N2P * 2);
    float*    b2p  = (float*)   carve((size_t)N2P * 4);
    ushort_t* x1bf = (ushort_t*)carve((size_t)M1 * HID * 2);
    float*    xw1t = (float*)   carve((size_t)T_SEQ * BATCH * N1 * 4);
    float*    xw2  = (float*)   carve((size_t)M1 * N2P * 4);
    (void)ws_size; (void)in_sizes; (void)n_in; (void)out_size;

    // 1) pack / convert
    {
        long long n = (long long)M1 * KP1;
        pack_x_kernel<<<(unsigned)((n + 255) / 256), 256, 0, stream>>>(imgs, lbls, X);
    }
    pack_w1_kernel<<<(KP1 * N1 + 255) / 256, 256, 0, stream>>>(W1, Wp1);
    pack_u1_kernel<<<(128 * 16 * 32 * 16 + 255) / 256, 256, 0, stream>>>(U1, Up1);
    pack_w2_kernel<<<(HID * N2P + 255) / 256, 256, 0, stream>>>(W2, Wp2);
    pack_b2_kernel<<<1, N2P, 0, stream>>>(b2, b2p);

    // 2) xw1 = seq @ W1 + b1   (stored t-major for the recurrent kernel)
    gemm_bf16_kernel<<<dim3(M1 / 64, N1 / 64), 128, 0, stream>>>(
        X, Wp1, b1, xw1t, KP1, KP1, N1, N1, /*remap=*/1);

    // 3) layer-1 recurrence (persistent, batch-parallel)
    lstm1_kernel<<<BATCH / 16, 256, 0, stream>>>(Up1, xw1t, x1bf);

    // 4) xw2 = x @ W2 + b2
    gemm_bf16_kernel<<<dim3(M1 / 64, N2P / 64), 128, 0, stream>>>(
        x1bf, Wp2, b2p, xw2, HID, HID, N2P, N2P, /*remap=*/0);

    // 5) layer-2 recurrence -> d_out
    lstm2_kernel<<<BATCH / 128, 128, 0, stream>>>(xw2, U2, out);
}
